// Summary_62594853372413
// MI455X (gfx1250) — matrix operand order repaired and checked by reading, not compiled, not run on hardware
//
#include <hip/hip_runtime.h>

typedef __attribute__((ext_vector_type(16))) _Float16 v16h;
typedef __attribute__((ext_vector_type(8)))  _Float16 v8h;
typedef __attribute__((ext_vector_type(4)))  _Float16 v4h;
typedef __attribute__((ext_vector_type(8)))  float    v8f;
typedef __attribute__((ext_vector_type(4)))  float    v4f;

#define N_NODES 8192
#define N_EDGES 524288
#define EMBED   128
#define HIDDEN  256
#define KDIM12  384   /* EMBED + 2*EMBED : fused GEMM1+GEMM2 reduction dim */
#define KDIM3   512   /* 2*HIDDEN */

/* workspace layout (bytes, all 256-aligned) */
#define OFF_X   0u          /* f16 [8192][384] fused activations  : 6,291,456 B */
#define OFF_H   6291456u    /* f16 [8192][256] hidden (relu)      : 4,194,304 B */
#define OFF_WC  10485760u   /* f16 [256][384]  [W_t | W_n]        :   196,608 B */
#define OFF_WR  10682368u   /* f16 [256][512]  W_r                :   262,144 B */
#define OFF_BC  10944512u   /* f32 [256]       b_t + b_n          :     1,024 B */

/* ---- weight fold + f16 conversion: 897*256 = 229,632 threads exactly ---- */
__global__ void k_prep(const float* __restrict__ Wt, const float* __restrict__ Wn,
                       const float* __restrict__ Wr, const float* __restrict__ bt,
                       const float* __restrict__ bn,
                       _Float16* __restrict__ Wc16, _Float16* __restrict__ Wr16,
                       float* __restrict__ bc) {
  const int idx = blockIdx.x * 256 + threadIdx.x;
  if (idx < HIDDEN * KDIM12) {
    const int j = idx / KDIM12;
    const int k = idx - j * KDIM12;
    const float w = (k < EMBED) ? Wt[j * EMBED + k]
                                : Wn[j * (2 * EMBED) + (k - EMBED)];
    Wc16[idx] = (_Float16)w;
  } else if (idx < HIDDEN * KDIM12 + HIDDEN * KDIM3) {
    const int i = idx - HIDDEN * KDIM12;
    Wr16[i] = (_Float16)Wr[i];
  } else {
    const int i = idx - HIDDEN * KDIM12 - HIDDEN * KDIM3;
    bc[i] = bt[i] + bn[i];
  }
}

/* ---- one wave32 per node: self gather + ragged segment-mean ----
 * Bag n = edges [offsets[n], offsets[n+1]) (last bag ends at N_EDGES),
 * which reproduces searchsorted(right)-1 semantics incl. empty bags.
 * Each lane owns a float4 slice of the 128-float row -> full 512B
 * coalesced row load per wave. 1024 blocks * 8 waves = 8192 nodes.      */
__global__ void k_gather(const int* __restrict__ entities,
                         const int* __restrict__ ne,
                         const int* __restrict__ nr,
                         const int* __restrict__ offsets,
                         const float* __restrict__ entity_emb,
                         const float* __restrict__ relation_emb,
                         _Float16* __restrict__ X) {
  const int lane = threadIdx.x & 31;
  const int wave = threadIdx.x >> 5;
  const int node = blockIdx.x * 8 + wave;

  const int start = offsets[node];
  const int end   = (node + 1 < N_NODES) ? offsets[node + 1] : N_EDGES;
  const int c4    = lane * 4;

  v4f acc_e = {0.f, 0.f, 0.f, 0.f};
  v4f acc_r = {0.f, 0.f, 0.f, 0.f};
  for (int e = start; e < end; ++e) {
    const float* pe = entity_emb   + (size_t)ne[e] * EMBED;
    const float* pr = relation_emb + (size_t)nr[e] * EMBED;
    if (e + 1 < end)  /* uniform per wave -> no divergence */
      __builtin_prefetch(entity_emb + (size_t)ne[e + 1] * EMBED + c4, 0, 0);
    acc_e += *(const v4f*)(pe + c4);
    acc_r += *(const v4f*)(pr + c4);
  }
  const int   cnt = end - start;
  const float inv = 1.0f / (float)((cnt > 0) ? cnt : 1);
  const v4f   invv = {inv, inv, inv, inv};

  const v4f self = *(const v4f*)(entity_emb + (size_t)entities[node] * EMBED + c4);

  _Float16* xr = X + (size_t)node * KDIM12;
  *(v4h*)(xr + c4)               = __builtin_convertvector(self, v4h);
  *(v4h*)(xr + EMBED + c4)       = __builtin_convertvector(acc_e * invv, v4h);
  *(v4h*)(xr + 2 * EMBED + c4)   = __builtin_convertvector(acc_r * invv, v4h);
}

/* ---- WMMA tile helpers: fragments per ISA 7.12.2 VGPR striping ----
 * A 16x32 f16: lane<16 holds row (lane&15), K {k0+0..7, k0+16..23};
 *              lane>=16 K {k0+8..15, k0+24..31}  -> two 16B loads.
 * B 32x16 f16 from row-major W[N][K] (B[k][n]=W[n][k]): lane group
 * half=lane>>4 holds K {k0+8*half..+7, k0+16+8*half..+7} -> two 16B loads. */
union frag16 { v16h v; v8h h[2]; };

__device__ __forceinline__ v8f wmma_rowcol(const _Float16* __restrict__ Arow,
                                           const _Float16* __restrict__ Brow,
                                           int K, int half) {
  v8f c = {0.f, 0.f, 0.f, 0.f, 0.f, 0.f, 0.f, 0.f};
  for (int k0 = 0; k0 < K; k0 += 32) {
    frag16 a, b;
    a.h[0] = *(const v8h*)(Arow + k0 + 8 * half);
    a.h[1] = *(const v8h*)(Arow + k0 + 16 + 8 * half);
    b.h[0] = *(const v8h*)(Brow + k0 + 8 * half);
    b.h[1] = *(const v8h*)(Brow + k0 + 16 + 8 * half);
    c = __builtin_amdgcn_wmma_f32_16x16x32_f16(false, a.v, false, b.v,
                                               (short)0, c, false, false);
  }
  return c;
}

/* ---- fused GEMM1+GEMM2: H = relu(X[8192x384] @ Wc.T + bc), f16 out ----
 * 8192x256 output = 512x16 wave-tiles = 1024 blocks * 8 waves exactly.  */
__global__ void k_gemm12(const _Float16* __restrict__ X,
                         const _Float16* __restrict__ Wc16,
                         const float* __restrict__ bc,
                         _Float16* __restrict__ H) {
  const int lane = threadIdx.x & 31;
  const int wave = threadIdx.x >> 5;
  const int tile = blockIdx.x * 8 + wave;
  const int tn = tile & 15;          /* 256/16 N-tiles */
  const int tm = tile >> 4;
  const int half = lane >> 4;
  const int l16  = lane & 15;

  const _Float16* Arow = X    + (size_t)(tm * 16 + l16) * KDIM12;
  const _Float16* Brow = Wc16 + (size_t)(tn * 16 + l16) * KDIM12;
  v8f c = wmma_rowcol(Arow, Brow, KDIM12, half);

  const int   col  = tn * 16 + l16;
  const float bias = bc[col];
#pragma unroll
  for (int v = 0; v < 8; ++v) {      /* D layout: M = v + 8*half, N = l16 */
    const int row = tm * 16 + v + 8 * half;
    const float r = fmaxf(c[v] + bias, 0.0f);
    H[(size_t)row * HIDDEN + col] = (_Float16)r;
  }
}

/* ---- GEMM3: pair = relu(H[4096x512] @ Wr.T + br), f32 out ----
 * 4096x256 output = 256x16 wave-tiles = 512 blocks * 8 waves exactly.   */
__global__ void k_gemm3(const _Float16* __restrict__ H,
                        const _Float16* __restrict__ Wr16,
                        const float* __restrict__ br,
                        float* __restrict__ out) {
  const int lane = threadIdx.x & 31;
  const int wave = threadIdx.x >> 5;
  const int tile = blockIdx.x * 8 + wave;
  const int tn = tile & 15;
  const int tm = tile >> 4;
  const int half = lane >> 4;
  const int l16  = lane & 15;

  const _Float16* Arow = H    + (size_t)(tm * 16 + l16) * KDIM3;
  const _Float16* Brow = Wr16 + (size_t)(tn * 16 + l16) * KDIM3;
  v8f c = wmma_rowcol(Arow, Brow, KDIM3, half);

  const int   col  = tn * 16 + l16;
  const float bias = br[col];
#pragma unroll
  for (int v = 0; v < 8; ++v) {
    const int row = tm * 16 + v + 8 * half;
    out[(size_t)row * HIDDEN + col] = fmaxf(c[v] + bias, 0.0f);
  }
}

extern "C" void kernel_launch(void* const* d_in, const int* in_sizes, int n_in,
                              void* d_out, int out_size, void* d_ws, size_t ws_size,
                              hipStream_t stream) {
  const int*   entities     = (const int*)d_in[0];
  const int*   ne           = (const int*)d_in[1];
  const int*   nr           = (const int*)d_in[2];
  const int*   offsets      = (const int*)d_in[3];
  const float* entity_emb   = (const float*)d_in[4];
  const float* relation_emb = (const float*)d_in[5];
  const float* Wt = (const float*)d_in[6];
  const float* bt = (const float*)d_in[7];
  const float* Wn = (const float*)d_in[8];
  const float* bn = (const float*)d_in[9];
  const float* Wr = (const float*)d_in[10];
  const float* br = (const float*)d_in[11];

  char* ws = (char*)d_ws;
  _Float16* X    = (_Float16*)(ws + OFF_X);
  _Float16* H    = (_Float16*)(ws + OFF_H);
  _Float16* Wc16 = (_Float16*)(ws + OFF_WC);
  _Float16* Wr16 = (_Float16*)(ws + OFF_WR);
  float*    bc   = (float*)(ws + OFF_BC);

  k_prep  <<<897,  256, 0, stream>>>(Wt, Wn, Wr, bt, bn, Wc16, Wr16, bc);
  k_gather<<<1024, 256, 0, stream>>>(entities, ne, nr, offsets,
                                     entity_emb, relation_emb, X);
  k_gemm12<<<1024, 256, 0, stream>>>(X, Wc16, bc, H);
  k_gemm3 <<<512,  256, 0, stream>>>(H, Wr16, br, (float*)d_out);
}
